// ExpressionDWTLoss_17480516895397
// MI455X (gfx1250) — matrix operand order repaired and checked by reading, not compiled, not run on hardware
//
#include <hip/hip_runtime.h>
#include <hip/hip_bf16.h>

// ---------------------------------------------------------------------------
// soft-DTW (gamma = 0.1) for x:[B,1024,64], y:[B,1024,64] fp32 -> out:[B]
//   Phase 1: S = x @ y^T per batch via V_WMMA_F32_16X16X32_BF16 (bf16x3 split
//            for ~fp32 accuracy), written to d_ws (fits in 192MB L2).
//   Phase 2: anti-diagonal wavefront DP, one 1024-thread workgroup per batch,
//            LDS double-buffered diagonals, softmin via fast exp/log.
// ---------------------------------------------------------------------------

#define SDTW_N 1024
#define SDTW_M 1024
#define SDTW_K 64
#define SDTW_GAMMA 0.1f
#define SDTW_INVG  10.0f
#define SDTW_BIG   1e10f

typedef __attribute__((ext_vector_type(16))) __bf16 v16bf;
typedef __attribute__((ext_vector_type(8)))  float  v8f;

// ---------------------------------------------------------------------------
// Phase 1: batched GEMM  S[g, i, j] = dot(x[b0+g, i, :], y[b0+g, j, :])
// 256 threads = 8 waves per block; each wave computes one 16x16 tile.
// Tiles per batch: (1024/16)^2 = 4096. Grid.x = g * 512.
// ---------------------------------------------------------------------------
__global__ __launch_bounds__(256) void sdtw_gemm_wmma(
    const float* __restrict__ x, const float* __restrict__ y,
    float* __restrict__ S, int b0)
{
    const int lane = threadIdx.x & 31;
    const int wv   = threadIdx.x >> 5;
    const int t    = blockIdx.x * 8 + wv;       // global tile index in group
    const int gb   = t >> 12;                   // batch within group (4096 tiles/batch)
    const int rem  = t & 4095;
    const int ti   = rem >> 6;                  // 0..63 tile row
    const int tj   = rem & 63;                  // 0..63 tile col
    const int b    = b0 + gb;

    const int l    = lane & 15;
    const int half = lane >> 4;

    const float* xb = x + (size_t)b * SDTW_N * SDTW_K;
    const float* yb = y + (size_t)b * SDTW_M * SDTW_K;
    // A: lanes 0-15 hold M=lane, lanes 16-31 hold M=lane-16 (half selects k-range)
    const float* arow = xb + (size_t)(ti * 16 + l) * SDTW_K;
    // B: lanes 0-15 hold N=lane (k 0-7,16-23), lanes 16-31 hold N=lane-16 (k 8-15,24-31)
    const float* brow = yb + (size_t)(tj * 16 + l) * SDTW_K;

    v8f c = {};
#pragma unroll
    for (int kf = 0; kf < 2; ++kf) {            // K = 64 = 2 fragments of 32
        v16bf ah, al, bh, bl;
#pragma unroll
        for (int e = 0; e < 16; ++e) {
            // A 16x32 bf16 layout: k = (e<8 ? e : e+8) + 8*half
            const int ka = ((e < 8) ? e : e + 8) + half * 8 + kf * 32;
            const float xv = arow[ka];
            const __bf16 xh = (__bf16)xv;
            ah[e] = xh;
            al[e] = (__bf16)(xv - (float)xh);
            // B 32x16 bf16 layout: k = (e<8 ? e : e+8) + 8*half
            const int kb = ((e < 8) ? e : e + 8) + half * 8 + kf * 32;
            const float yv = brow[kb];
            const __bf16 yh = (__bf16)yv;
            bh[e] = yh;
            bl[e] = (__bf16)(yv - (float)yh);
        }
        // bf16x3: hi*hi + hi*lo + lo*hi  (lo*lo term ~2^-18 rel, negligible)
        c = __builtin_amdgcn_wmma_f32_16x16x32_bf16(false, ah, false, bh,
                                                    (short)0, c, false, false);
        c = __builtin_amdgcn_wmma_f32_16x16x32_bf16(false, ah, false, bl,
                                                    (short)0, c, false, false);
        c = __builtin_amdgcn_wmma_f32_16x16x32_bf16(false, al, false, bh,
                                                    (short)0, c, false, false);
    }

    // C/D layout: N = lane&15, M = r + 8*(lane>=16)
    float* Sb = S + (size_t)gb * SDTW_N * SDTW_M;
#pragma unroll
    for (int r = 0; r < 8; ++r) {
        const int row = ti * 16 + r + half * 8;
        const int col = tj * 16 + l;
        Sb[(size_t)row * SDTW_M + col] = c[r];
    }
}

// ---------------------------------------------------------------------------
// softmin_gamma(a,b,c) = m - gamma*log(sum exp((m - .)/gamma)), m = min
// ---------------------------------------------------------------------------
__device__ __forceinline__ float softmin3(float a, float b, float c)
{
    const float m = fminf(a, fminf(b, c));
    const float s = __expf((m - a) * SDTW_INVG) +
                    __expf((m - b) * SDTW_INVG) +
                    __expf((m - c) * SDTW_INVG);
    return m - SDTW_GAMMA * __logf(s);
}

// ---------------------------------------------------------------------------
// Phase 2: wavefront DP. One block of 1024 threads per batch; thread = row i.
// Diagonal d holds cells (i, j=d-i). Two LDS diagonals, slot [i+1] = row i.
// ---------------------------------------------------------------------------
__global__ __launch_bounds__(1024) void sdtw_dp(
    const float* __restrict__ x, const float* __restrict__ y,
    const float* __restrict__ S, float* __restrict__ out, int b0)
{
    const int b = b0 + blockIdx.x;
    const int i = threadIdx.x;

    __shared__ float y2s[SDTW_M];
    __shared__ float bufA[SDTW_N + 1];
    __shared__ float bufB[SDTW_N + 1];

    // Row norms: x2 kept in a register, y2 shared across the wavefront.
    const float* xr = x + ((size_t)b * SDTW_N + i) * SDTW_K;
    const float* yr = y + ((size_t)b * SDTW_M + i) * SDTW_K;
    float x2 = 0.f, y2 = 0.f;
#pragma unroll 8
    for (int k = 0; k < SDTW_K; ++k) { x2 = fmaf(xr[k], xr[k], x2); }
#pragma unroll 8
    for (int k = 0; k < SDTW_K; ++k) { y2 = fmaf(yr[k], yr[k], y2); }
    y2s[i] = y2;

    const float* Srow = S + ((size_t)blockIdx.x * SDTW_N + i) * SDTW_M;

    // init: diag(-1) = BIG everywhere; diag(0) = BIG except R[0,0] = D(0,0)
    bufA[i + 1] = SDTW_BIG;
    bufB[i + 1] = (i == 0) ? (x2 + y2 - 2.f * Srow[0]) : SDTW_BIG;
    if (i == 0) { bufA[0] = SDTW_BIG; bufB[0] = SDTW_BIG; }
    __syncthreads();

    float* r2 = bufA;   // diag d-2
    float* r1 = bufB;   // diag d-1
    float rnew = SDTW_BIG;

    for (int d = 1; d <= SDTW_N + SDTW_M - 2; ++d) {
        const int j = d - i;
        const float v11 = r2[i];       // R[i-1, j-1]
        const float v10 = r1[i];       // R[i-1, j  ]
        const float v01 = r1[i + 1];   // R[i,   j-1]
        rnew = SDTW_BIG;
        if (j >= 0 && j < SDTW_M) {
            const float Dv = x2 + y2s[j] - 2.f * Srow[j];
            rnew = Dv + softmin3(v11, v10, v01);
        }
        __syncthreads();               // all reads of r2 done
        r2[i + 1] = rnew;              // r2 buffer now holds diag d
        float* tswap = r1; r1 = r2; r2 = tswap;
        __syncthreads();               // diag d visible to neighbors
    }

    if (i == SDTW_N - 1) out[b] = rnew;   // R[N-1, M-1]
}

// ---------------------------------------------------------------------------
// Fallback (only if workspace can't hold even one batch of S): fused DP with
// inline dot products from global memory. Slow but correct.
// ---------------------------------------------------------------------------
__global__ __launch_bounds__(1024) void sdtw_dp_fused(
    const float* __restrict__ x, const float* __restrict__ y,
    float* __restrict__ out)
{
    const int b = blockIdx.x;
    const int i = threadIdx.x;

    __shared__ float y2s[SDTW_M];
    __shared__ float bufA[SDTW_N + 1];
    __shared__ float bufB[SDTW_N + 1];

    const float* xr = x + ((size_t)b * SDTW_N + i) * SDTW_K;
    const float* yb = y + (size_t)b * SDTW_M * SDTW_K;
    const float* yr = yb + (size_t)i * SDTW_K;
    float x2 = 0.f, y2 = 0.f;
#pragma unroll 8
    for (int k = 0; k < SDTW_K; ++k) { x2 = fmaf(xr[k], xr[k], x2); }
#pragma unroll 8
    for (int k = 0; k < SDTW_K; ++k) { y2 = fmaf(yr[k], yr[k], y2); }
    y2s[i] = y2;

    float d00 = 0.f;
    if (i == 0) {
        float dot = 0.f;
        for (int k = 0; k < SDTW_K; ++k) dot = fmaf(xr[k], yb[k], dot);
        d00 = x2 + y2 - 2.f * dot;
    }
    bufA[i + 1] = SDTW_BIG;
    bufB[i + 1] = (i == 0) ? d00 : SDTW_BIG;
    if (i == 0) { bufA[0] = SDTW_BIG; bufB[0] = SDTW_BIG; }
    __syncthreads();

    float* r2 = bufA;
    float* r1 = bufB;
    float rnew = SDTW_BIG;

    for (int d = 1; d <= SDTW_N + SDTW_M - 2; ++d) {
        const int j = d - i;
        const float v11 = r2[i];
        const float v10 = r1[i];
        const float v01 = r1[i + 1];
        rnew = SDTW_BIG;
        if (j >= 0 && j < SDTW_M) {
            const float* yrj = yb + (size_t)j * SDTW_K;
            float dot = 0.f;
#pragma unroll 8
            for (int k = 0; k < SDTW_K; ++k) dot = fmaf(xr[k], yrj[k], dot);
            const float Dv = x2 + y2s[j] - 2.f * dot;
            rnew = Dv + softmin3(v11, v10, v01);
        }
        __syncthreads();
        r2[i + 1] = rnew;
        float* tswap = r1; r1 = r2; r2 = tswap;
        __syncthreads();
    }

    if (i == SDTW_N - 1) out[b] = rnew;
}

// ---------------------------------------------------------------------------
extern "C" void kernel_launch(void* const* d_in, const int* in_sizes, int n_in,
                              void* d_out, int out_size, void* d_ws, size_t ws_size,
                              hipStream_t stream) {
    const float* x = (const float*)d_in[0];
    const float* y = (const float*)d_in[1];
    float* out = (float*)d_out;
    float* S   = (float*)d_ws;

    const int B = in_sizes[0] / (SDTW_N * SDTW_K);   // 32

    const size_t bytesPerBatch = (size_t)SDTW_N * SDTW_M * sizeof(float); // 4 MB
    int G = (int)(ws_size / bytesPerBatch);
    if (G > B) G = B;

    if (G >= 1) {
        for (int b0 = 0; b0 < B; b0 += G) {
            const int g = (B - b0 < G) ? (B - b0) : G;
            // 4096 tiles/batch, 8 tiles/block -> 512 blocks per batch
            sdtw_gemm_wmma<<<g * 512, 256, 0, stream>>>(x, y, S, b0);
            sdtw_dp<<<g, 1024, 0, stream>>>(x, y, S, out, b0);
        }
    } else {
        sdtw_dp_fused<<<B, 1024, 0, stream>>>(x, y, out);
    }
}
